// CAMixer_3246995275933
// MI455X (gfx1250) — matrix operand order repaired and checked by reading, not compiled, not run on hardware
//
#include <hip/hip_runtime.h>
#include <hip/hip_bf16.h>
#include <cstdint>
#include <cstddef>

// ---------------------------------------------------------------------------
// CDNA5 (gfx1250) CAMixer forward. All GEMM-like stages use
// v_wmma_f32_16x16x32_f16 (f16 operands, f32 accumulate), wave32.
// Fragments are fed by vectorized b128 loads (global f16 / LDS f16 staging).
// ---------------------------------------------------------------------------

typedef __attribute__((ext_vector_type(16))) _Float16 v16h;
typedef __attribute__((ext_vector_type(8)))  _Float16 v8h;
typedef __attribute__((ext_vector_type(4)))  _Float16 v4h;
typedef __attribute__((ext_vector_type(8)))  float    v8f;
typedef __attribute__((ext_vector_type(4)))  float    v4f;

#define WMMA16(a, b, c) \
  __builtin_amdgcn_wmma_f32_16x16x32_f16(false, (a), false, (b), (short)0, (c), false, false)
#define SHUF16(lo, hi) \
  __builtin_shufflevector((lo), (hi), 0, 1, 2, 3, 4, 5, 6, 7, 8, 9, 10, 11, 12, 13, 14, 15)

// WMMA f16 operand k order used here (contiguous; A and B match, not ISA table order), wave32:
//  A (16x32, MxK): lane holds row M=lane&15; half hi = K (lane>=16)*16 + hi.
//                  -> one contiguous 32B chunk in row-major f16.
//  B (32x16, KxN): lane holds col N=lane&15; half hi = K (lane>=16)*16 + hi.
//                  -> one contiguous 32B chunk in K-major f16.
__device__ __forceinline__ int a_koff(int lane, int v) {
  return ((lane >> 4) << 4) + (v << 1);
}
__device__ __forceinline__ int b_koff(int lane, int hi) {
  return ((lane >> 4) << 4) + hi;
}
__device__ __forceinline__ float leaky02(float x) { return x >= 0.f ? x : 0.2f * x; }

#define PP   65536   // H*W
#define CC   128
#define BB   4

// ---------------------------------------------------------------------------
// 1x1-conv GEMM over NCHW: Y[b,co,p] = sum_ci W[co,ci]*X[b,ci,p] + bias[co]
// grid(P/64, B), block 256 (8 waves). W and transposed X tile staged in LDS
// as f16 once per block; fragments are pure ds_load_b128.
// ---------------------------------------------------------------------------
__global__ void __launch_bounds__(256)
gemm128_kernel(const float* __restrict__ W, const float* __restrict__ bias,
               const float* __restrict__ X, float* __restrict__ Y) {
  __shared__ _Float16 lw[128][128];  // lw[m][k]   32 KB
  __shared__ _Float16 lx[64][128];   // lx[n][k]   16 KB (transposed tile)
  const int tid = threadIdx.x;
  const int lane = tid & 31, wv = tid >> 5;
  const int ml = lane & 15, lh = lane >> 4;
  const int b = blockIdx.y;
  const int p0 = blockIdx.x * 64;
  // stage W as f16
  for (int i = tid; i < 128 * 32; i += 256) {
    const int m = i >> 5, kg = (i & 31) << 2;
    const v4f w4 = *(const v4f*)(W + m * 128 + kg);
    v4h h;
    h[0] = (_Float16)w4[0]; h[1] = (_Float16)w4[1];
    h[2] = (_Float16)w4[2]; h[3] = (_Float16)w4[3];
    *(v4h*)&lw[m][kg] = h;
  }
  // stage X tile transposed (k-major rows for B fragments)
  for (int i = tid; i < 128 * 16; i += 256) {
    const int k = i >> 4, ng = (i & 15) << 2;
    const v4f x4 = *(const v4f*)(X + (((size_t)b * 128 + k) << 16) + p0 + ng);
    lx[ng + 0][k] = (_Float16)x4[0];
    lx[ng + 1][k] = (_Float16)x4[1];
    lx[ng + 2][k] = (_Float16)x4[2];
    lx[ng + 3][k] = (_Float16)x4[3];
  }
  __syncthreads();
  const int m0 = wv * 16;
  v8f acc[4] = {};
  for (int ks = 0; ks < 4; ++ks) {
    const int ab = ks * 32 + (lh << 4);
    const v8h alo = *(const v8h*)&lw[m0 + ml][ab];
    const v8h ahi = *(const v8h*)&lw[m0 + ml][ab + 8];
    const v16h af = SHUF16(alo, ahi);
#pragma unroll
    for (int nt = 0; nt < 4; ++nt) {
      const v16h bf = *(const v16h*)&lx[nt * 16 + ml][ks * 32 + (lh << 4)];
      acc[nt] = WMMA16(af, bf, acc[nt]);
    }
  }
#pragma unroll
  for (int nt = 0; nt < 4; ++nt)
#pragma unroll
    for (int r = 0; r < 8; ++r) {
      const int m = m0 + r + (lh << 3);
      const int n = p0 + nt * 16 + ml;
      Y[(((size_t)b * 128 + m) << 16) + n] = acc[nt][r] + bias[m];
    }
}

// ---------------------------------------------------------------------------
// rin: f[b,o,p] = leaky( sum_{ci<131} rin_w[o,ci]*cond[b,ci,p] + rin_b[o] )
// cond = [v(128) | condition_global(1) | cy | cx].  K padded 131 -> 160.
// grid(P/16, B), block 64 (2 waves; wave = M tile of 16)
// ---------------------------------------------------------------------------
__device__ __forceinline__ float cond_val(const float* __restrict__ v,
                                          const float* __restrict__ cg,
                                          int b, int ci, int p) {
  if (ci < 128) return v[(((size_t)b * 128 + ci) << 16) + p];
  if (ci == 128) return cg[((size_t)b << 16) + p];
  if (ci == 129) return -1.f + (2.f / 15.f) * (float)((p >> 8) & 15);
  if (ci == 130) return -1.f + (2.f / 15.f) * (float)(p & 15);
  return 0.f;
}

__global__ void __launch_bounds__(64)
rin_kernel(const float* __restrict__ rw, const float* __restrict__ rb,
           const float* __restrict__ v, const float* __restrict__ cg,
           float* __restrict__ f) {
  const int lane = threadIdx.x & 31, wv = threadIdx.x >> 5;  // wv: M tile 0..1
  const int ml = lane & 15, lh = lane >> 4;
  const int b = blockIdx.y;
  const int p0 = blockIdx.x * 16;
  const int m0 = wv * 16;
  const int n = p0 + ml;
  v8f acc = {};
  for (int ks = 0; ks < 5; ++ks) {
    const int k0 = ks * 32;
    v16h af;
    const int row = m0 + ml;
#pragma unroll
    for (int vv = 0; vv < 8; ++vv) {
      int k = k0 + a_koff(lane, vv);
      af[2 * vv] = (_Float16)((k < 131) ? rw[row * 131 + k] : 0.f);
      af[2 * vv + 1] = (_Float16)((k + 1 < 131) ? rw[row * 131 + k + 1] : 0.f);
    }
    v16h bf;
#pragma unroll
    for (int hi = 0; hi < 16; ++hi) {
      int ci = k0 + b_koff(lane, hi);
      bf[hi] = (_Float16)((ci < 131) ? cond_val(v, cg, b, ci, n) : 0.f);
    }
    acc = WMMA16(af, bf, acc);
  }
#pragma unroll
  for (int r = 0; r < 8; ++r) {
    const int co = m0 + r + (lh << 3);
    f[(((size_t)b * 32 + co) << 16) + n] = leaky02(acc[r] + rb[co]);
  }
}

// ---------------------------------------------------------------------------
// offsets (roff1->leaky->roff2) + channel-mean of f ; one thread per (b,p)
// ---------------------------------------------------------------------------
__global__ void offs_kernel(const float* __restrict__ f,
                            const float* __restrict__ w1, const float* __restrict__ b1,
                            const float* __restrict__ w2, const float* __restrict__ b2,
                            float* __restrict__ offs, float* __restrict__ mchan) {
  const int idx = blockIdx.x * 256 + threadIdx.x;  // 0 .. 4*65536-1
  const int b = idx >> 16, p = idx & 65535;
  float fv[32], s = 0.f;
#pragma unroll
  for (int ci = 0; ci < 32; ++ci) {
    fv[ci] = f[(((size_t)b * 32 + ci) << 16) + p];
    s += fv[ci];
  }
  mchan[idx] = s * (1.f / 32.f);
  float t[16];
#pragma unroll
  for (int o = 0; o < 16; ++o) {
    float a = b1[o];
#pragma unroll
    for (int ci = 0; ci < 32; ++ci) a += w1[o * 32 + ci] * fv[ci];
    t[o] = leaky02(a);
  }
#pragma unroll
  for (int o = 0; o < 2; ++o) {
    float a = b2[o];
#pragma unroll
    for (int i = 0; i < 16; ++i) a += w2[o * 16 + i] * t[i];
    offs[(((size_t)b * 2 + o) << 16) + p] = a;
  }
}

// ---------------------------------------------------------------------------
// fmean: per (b,ci) spatial mean of f.  grid 128, block 256.
// ---------------------------------------------------------------------------
__global__ void fmean_kernel(const float* __restrict__ f, float* __restrict__ fmean) {
  __shared__ float sm[256];
  const int bci = blockIdx.x;
  float s = 0.f;
  for (int i = threadIdx.x; i < 65536; i += 256) s += f[((size_t)bci << 16) + i];
  sm[threadIdx.x] = s;
  __syncthreads();
  for (int st = 128; st > 0; st >>= 1) {
    if ((int)threadIdx.x < st) sm[threadIdx.x] += sm[threadIdx.x + st];
    __syncthreads();
  }
  if (threadIdx.x == 0) fmean[bci] = sm[0] * (1.f / 65536.f);
}

__global__ void ca_kernel(const float* __restrict__ fmean, const float* __restrict__ w,
                          const float* __restrict__ bias, float* __restrict__ ca) {
  const int i = blockIdx.x * 256 + threadIdx.x;
  if (i >= 512) return;
  const int b = i >> 7, co = i & 127;
  float s = bias[co];
#pragma unroll
  for (int ci = 0; ci < 32; ++ci) s += w[co * 32 + ci] * fmean[b * 32 + ci];
  ca[i] = 1.f / (1.f + __expf(-s));
}

// ---------------------------------------------------------------------------
// sa = sigmoid(conv3x3(f, rsa_w) + b), pad 1.  one thread per (b,p)
// ---------------------------------------------------------------------------
__global__ void sa_kernel(const float* __restrict__ f, const float* __restrict__ w,
                          const float* __restrict__ bias, float* __restrict__ sa) {
  const int idx = blockIdx.x * 256 + threadIdx.x;
  const int b = idx >> 16, p = idx & 65535;
  const int h = p >> 8, wc = p & 255;
  float s = bias[0];
  for (int ci = 0; ci < 32; ++ci) {
    const size_t base = ((size_t)b * 32 + ci) << 16;
#pragma unroll
    for (int ky = 0; ky < 3; ++ky) {
      const int yy = h - 1 + ky;
      if ((unsigned)yy >= 256u) continue;
#pragma unroll
      for (int kx = 0; kx < 3; ++kx) {
        const int xx = wc - 1 + kx;
        if ((unsigned)xx >= 256u) continue;
        s += w[ci * 9 + ky * 3 + kx] * f[base + yy * 256 + xx];
      }
    }
  }
  sa[idx] = 1.f / (1.f + __expf(-s));
}

// ---------------------------------------------------------------------------
// mask MLP + hard gumbel (forward: mask = argmax(pred + g)==0). one thr/window
// ---------------------------------------------------------------------------
__global__ void mask_kernel(const float* __restrict__ mchan, const float* __restrict__ gum,
                            const float* __restrict__ rm1w, const float* __restrict__ rm1b,
                            const float* __restrict__ rm2w, const float* __restrict__ rm2b,
                            float* __restrict__ mask) {
  const int i = blockIdx.x * 256 + threadIdx.x;
  if (i >= 1024) return;
  const int b = i >> 8, n = i & 255, wy = n >> 4, wx = n & 15;
  float h1[16];
#pragma unroll
  for (int o = 0; o < 16; ++o) h1[o] = rm1b[o];
  for (int dh = 0; dh < 16; ++dh)
    for (int dw = 0; dw < 16; ++dw) {
      const int e = dh * 16 + dw;
      const float val = mchan[((size_t)b << 16) + (wy * 16 + dh) * 256 + wx * 16 + dw];
#pragma unroll
      for (int o = 0; o < 16; ++o) h1[o] += rm1w[o * 256 + e] * val;
    }
  float z0 = rm2b[0], z1 = rm2b[1];
#pragma unroll
  for (int o = 0; o < 16; ++o) {
    const float hv = leaky02(h1[o]);
    z0 += rm2w[o] * hv;
    z1 += rm2w[16 + o] * hv;
  }
  const float m = fmaxf(z0, z1);
  float p0 = __expf(z0 - m), p1 = __expf(z1 - m);
  const float inv = 1.f / (p0 + p1);
  p0 *= inv; p1 *= inv;
  const float u0 = gum[i * 2], u1 = gum[i * 2 + 1];
  const float g0 = -logf(-logf(u0 + 1e-10f) + 1e-10f);
  const float g1 = -logf(-logf(u1 + 1e-10f) + 1e-10f);
  mask[i] = (p0 + g0 >= p1 + g1) ? 1.f : 0.f;
}

// ---------------------------------------------------------------------------
// flow warp (bilinear, border clamp). one thread per (b,p), loops channels.
// ---------------------------------------------------------------------------
__global__ void warp_kernel(const float* __restrict__ x, const float* __restrict__ offs,
                            float* __restrict__ xw) {
  const int idx = blockIdx.x * 256 + threadIdx.x;
  const int b = idx >> 16, p = idx & 65535;
  const int gy = p >> 8, gx = p & 255;
  const float fx = offs[(((size_t)b * 2 + 0) << 16) + p];
  const float fy = offs[(((size_t)b * 2 + 1) << 16) + p];
  const float sx = fminf(fmaxf((float)gx + fx, 0.f), 255.f);
  const float sy = fminf(fmaxf((float)gy + fy, 0.f), 255.f);
  const float x0f = floorf(sx), y0f = floorf(sy);
  const float wxf = sx - x0f, wyf = sy - y0f;
  const int x0 = (int)x0f, y0 = (int)y0f;
  const int x1 = min(x0 + 1, 255), y1 = min(y0 + 1, 255);
  const float w00 = (1.f - wxf) * (1.f - wyf), w01 = wxf * (1.f - wyf);
  const float w10 = (1.f - wxf) * wyf, w11 = wxf * wyf;
  const int i00 = y0 * 256 + x0, i01 = y0 * 256 + x1;
  const int i10 = y1 * 256 + x0, i11 = y1 * 256 + x1;
  for (int c = 0; c < 128; ++c) {
    const size_t base = ((size_t)b * 128 + c) << 16;
    xw[base + p] = x[base + i00] * w00 + x[base + i01] * w01 +
                   x[base + i10] * w10 + x[base + i11] * w11;
  }
}

// ---------------------------------------------------------------------------
// stage: gxw[win,t,c] = xw*mask (f16, token-major, feeds A fragments)
//        gvt[win,c,t] = v*mask  (f16, K-major, feeds B fragments of P@V)
// ---------------------------------------------------------------------------
__global__ void stage_kernel(const float* __restrict__ xw, const float* __restrict__ v,
                             const float* __restrict__ mask,
                             _Float16* __restrict__ gxw, _Float16* __restrict__ gvt) {
  const int idx = blockIdx.x * 256 + threadIdx.x;  // 1024*128*256
  const int t = idx & 255, c = (idx >> 8) & 127, win = idx >> 15;
  const int b = win >> 8, n = win & 255, wy = n >> 4, wx = n & 15;
  const int py = wy * 16 + (t >> 4), px = wx * 16 + (t & 15);
  const float mk = mask[win];
  const size_t src = (((size_t)b * 128 + c) << 16) + py * 256 + px;
  gvt[idx] = (_Float16)(v[src] * mk);                       // [win][c][t]
  gxw[((size_t)win << 15) + t * 128 + c] = (_Float16)(xw[src] * mk);  // [win][t][c]
}

// ---------------------------------------------------------------------------
// Fused q/k projection per window: g{q,k}[win,t,c] = gxw_win @ W^T + bias (f16).
// block = 1 window = 8 waves (wave -> 16-channel tile); both weight matrices
// staged as f16 in LDS; A fragments shared between the q and k WMMAs.
// ---------------------------------------------------------------------------
__global__ void __launch_bounds__(256)
projqk_kernel(const _Float16* __restrict__ gxw,
              const float* __restrict__ Wq, const float* __restrict__ bq,
              const float* __restrict__ Wk, const float* __restrict__ bk,
              _Float16* __restrict__ gq, _Float16* __restrict__ gk) {
  __shared__ _Float16 lwq[128][128];  // 32 KB
  __shared__ _Float16 lwk[128][128];  // 32 KB
  const int tid = threadIdx.x;
  const int lane = tid & 31, wv = tid >> 5;
  const int ml = lane & 15, lh = lane >> 4;
  const int win = blockIdx.x;
  for (int i = tid; i < 128 * 32; i += 256) {
    const int m = i >> 5, kg = (i & 31) << 2;
    const v4f q4 = *(const v4f*)(Wq + m * 128 + kg);
    const v4f k4 = *(const v4f*)(Wk + m * 128 + kg);
    v4h hq, hk;
    hq[0] = (_Float16)q4[0]; hq[1] = (_Float16)q4[1];
    hq[2] = (_Float16)q4[2]; hq[3] = (_Float16)q4[3];
    hk[0] = (_Float16)k4[0]; hk[1] = (_Float16)k4[1];
    hk[2] = (_Float16)k4[2]; hk[3] = (_Float16)k4[3];
    *(v4h*)&lwq[m][kg] = hq;
    *(v4h*)&lwk[m][kg] = hk;
  }
  __syncthreads();
  const int nc = wv * 16 + ml;  // output channel (B col / D col)
  const float biasq = bq[nc], biask = bk[nc];
  const _Float16* xb = gxw + ((size_t)win << 15);
  for (int tt = 0; tt < 16; ++tt) {
    const _Float16* arow = xb + (size_t)(tt * 16 + ml) * 128;
    v8f aq = {}, ak = {};
    for (int ks = 0; ks < 4; ++ks) {
      const int ab = ks * 32 + (lh << 4);
      const v8h alo = *(const v8h*)(arow + ab);
      const v8h ahi = *(const v8h*)(arow + ab + 8);
      const v16h af = SHUF16(alo, ahi);
      const v16h bfq = *(const v16h*)&lwq[nc][ks * 32 + (lh << 4)];
      const v16h bfk = *(const v16h*)&lwk[nc][ks * 32 + (lh << 4)];
      aq = WMMA16(af, bfq, aq);
      ak = WMMA16(af, bfk, ak);
    }
#pragma unroll
    for (int r = 0; r < 8; ++r) {
      const int mo = tt * 16 + r + (lh << 3);
      const size_t o = (((size_t)win << 8) + mo) * 128 + nc;
      gq[o] = (_Float16)(aq[r] + biasq);
      gk[o] = (_Float16)(ak[r] + biask);
    }
  }
}

// ---------------------------------------------------------------------------
// Windowed attention: S = q1 k1^T, softmax, f_attn = P v1; out = f_attn + v*sa*(1-m)
// block = 1 window = 8 waves; each wave owns 2 row-tiles of 16 tokens.
// All fragments are aligned vector loads; P staged through per-wave LDS strip.
// ---------------------------------------------------------------------------
__global__ void __launch_bounds__(256)
attn_kernel(const _Float16* __restrict__ gq, const _Float16* __restrict__ gk,
            const _Float16* __restrict__ gvt, const float* __restrict__ v,
            const float* __restrict__ sa, const float* __restrict__ mask,
            float* __restrict__ out) {
  __shared__ _Float16 plds[8][16][256];  // 64 KB, per-wave strips
  const int win = blockIdx.x;
  const int b = win >> 8, n = win & 255, wy = n >> 4, wx = n & 15;
  const int lane = threadIdx.x & 31, wv = threadIdx.x >> 5;
  const int ml = lane & 15, lh = lane >> 4;
  const float invm = 1.f - mask[win];
  const _Float16* qb = gq + ((size_t)win << 15);
  const _Float16* kb = gk + ((size_t)win << 15);
  const _Float16* vtb = gvt + ((size_t)win << 15);

  for (int rt = 0; rt < 2; ++rt) {
    const int m0 = wv * 32 + rt * 16;
    // ---- S = q1 @ k1^T for 16 rows x 256 keys ----
    v8f acc[16] = {};
    for (int ks = 0; ks < 4; ++ks) {
      const int k0 = ks * 32;
      const _Float16* qrow = qb + (size_t)(m0 + ml) * 128 + k0 + (lh << 4);
      const v8h alo = *(const v8h*)qrow;
      const v8h ahi = *(const v8h*)(qrow + 8);
      const v16h af = SHUF16(alo, ahi);
#pragma unroll
      for (int nt = 0; nt < 16; ++nt) {
        const v16h bf =
            *(const v16h*)(kb + (size_t)(nt * 16 + ml) * 128 + k0 + (lh << 4));
        acc[nt] = WMMA16(af, bf, acc[nt]);
      }
    }
    // ---- row softmax (rows live in 16-lane halves of each acc VGPR) ----
#pragma unroll
    for (int r = 0; r < 8; ++r) {
      float mx = -3.4e38f;
#pragma unroll
      for (int nt = 0; nt < 16; ++nt) mx = fmaxf(mx, acc[nt][r]);
#pragma unroll
      for (int off = 1; off < 16; off <<= 1) mx = fmaxf(mx, __shfl_xor(mx, off, 16));
      float s = 0.f;
#pragma unroll
      for (int nt = 0; nt < 16; ++nt) {
        const float e = __expf(acc[nt][r] - mx);
        acc[nt][r] = e;
        s += e;
      }
#pragma unroll
      for (int off = 1; off < 16; off <<= 1) s += __shfl_xor(s, off, 16);
      const float inv = 1.f / s;
#pragma unroll
      for (int nt = 0; nt < 16; ++nt) acc[nt][r] *= inv;
    }
    // ---- P -> per-wave LDS (D layout -> row-major, becomes next A operand) ----
#pragma unroll
    for (int nt = 0; nt < 16; ++nt)
#pragma unroll
      for (int r = 0; r < 8; ++r)
        plds[wv][r + (lh << 3)][nt * 16 + ml] = (_Float16)acc[nt][r];
    // ---- f_attn = P @ v1 ----
    v8f acc2[8] = {};
    for (int ks = 0; ks < 8; ++ks) {
      const int k0 = ks * 32;
      const v8h alo = *(const v8h*)&plds[wv][ml][k0 + (lh << 4)];
      const v8h ahi = *(const v8h*)&plds[wv][ml][k0 + (lh << 4) + 8];
      const v16h af = SHUF16(alo, ahi);
#pragma unroll
      for (int ct = 0; ct < 8; ++ct) {
        const v16h bf =
            *(const v16h*)(vtb + (size_t)(ct * 16 + ml) * 256 + k0 + (lh << 4));
        acc2[ct] = WMMA16(af, bf, acc2[ct]);
      }
    }
    // ---- epilogue: + v*sa*(1-mask), scatter to NCHW ----
#pragma unroll
    for (int ct = 0; ct < 8; ++ct)
#pragma unroll
      for (int r = 0; r < 8; ++r) {
        const int t = m0 + r + (lh << 3);
        const int c = ct * 16 + ml;
        const int py = wy * 16 + (t >> 4), px = wx * 16 + (t & 15);
        const size_t pix = ((size_t)py << 8) + px;
        const size_t vi = (((size_t)b * 128 + c) << 16) + pix;
        out[vi] = acc2[ct][r] + v[vi] * sa[((size_t)b << 16) + pix] * invm;
      }
  }
}

// ---------------------------------------------------------------------------
// depthwise 5x5 (pad, dilation).  one thread per output element.
// ---------------------------------------------------------------------------
__global__ void dw5_kernel(const float* __restrict__ X, const float* __restrict__ W,
                           const float* __restrict__ bias, float* __restrict__ Y,
                           int pad, int dil) {
  const int idx = blockIdx.x * 256 + threadIdx.x;  // 4*128*65536
  const int p = idx & 65535, c = (idx >> 16) & 127, b = idx >> 23;
  const int h = p >> 8, w = p & 255;
  float s = bias[c];
  const size_t base = ((size_t)b * 128 + c) << 16;
#pragma unroll
  for (int ky = 0; ky < 5; ++ky) {
    const int yy = h - pad + ky * dil;
    if ((unsigned)yy >= 256u) continue;
#pragma unroll
    for (int kx = 0; kx < 5; ++kx) {
      const int xx = w - pad + kx * dil;
      if ((unsigned)xx >= 256u) continue;
      s += W[c * 25 + ky * 5 + kx] * X[base + yy * 256 + xx];
    }
  }
  Y[idx] = s;
}

// out2 = gelu(cs) * ca[b,c] + residual
__global__ void out2_kernel(const float* __restrict__ cs, const float* __restrict__ resid,
                            const float* __restrict__ ca, float* __restrict__ y) {
  const int idx = blockIdx.x * 256 + threadIdx.x;
  const int c = (idx >> 16) & 127, b = idx >> 23;
  const float xv = cs[idx];
  const float g = 0.5f * xv * (1.f + erff(xv * 0.70710678118654752f));
  y[idx] = g * ca[b * 128 + c] + resid[idx];
}

// ---------------------------------------------------------------------------
extern "C" void kernel_launch(void* const* d_in, const int* in_sizes, int n_in,
                              void* d_out, int out_size, void* d_ws, size_t ws_size,
                              hipStream_t stream) {
  (void)in_sizes; (void)n_in; (void)out_size; (void)ws_size;
  const float* x     = (const float*)d_in[0];
  const float* cg    = (const float*)d_in[1];
  const float* gum   = (const float*)d_in[2];
  const float* pv_w  = (const float*)d_in[3];  const float* pv_b  = (const float*)d_in[4];
  const float* pq_w  = (const float*)d_in[5];  const float* pq_b  = (const float*)d_in[6];
  const float* pk_w  = (const float*)d_in[7];  const float* pk_b  = (const float*)d_in[8];
  const float* cs1_w = (const float*)d_in[9];  const float* cs1_b = (const float*)d_in[10];
  const float* cs2_w = (const float*)d_in[11]; const float* cs2_b = (const float*)d_in[12];
  const float* cs3_w = (const float*)d_in[13]; const float* cs3_b = (const float*)d_in[14];
  const float* po_w  = (const float*)d_in[15]; const float* po_b  = (const float*)d_in[16];
  const float* rin_w = (const float*)d_in[17]; const float* rin_b = (const float*)d_in[18];
  const float* ro1_w = (const float*)d_in[19]; const float* ro1_b = (const float*)d_in[20];
  const float* ro2_w = (const float*)d_in[21]; const float* ro2_b = (const float*)d_in[22];
  const float* rm1_w = (const float*)d_in[23]; const float* rm1_b = (const float*)d_in[24];
  const float* rm2_w = (const float*)d_in[25]; const float* rm2_b = (const float*)d_in[26];
  const float* rca_w = (const float*)d_in[27]; const float* rca_b = (const float*)d_in[28];
  const float* rsa_w = (const float*)d_in[29]; const float* rsa_b = (const float*)d_in[30];

  char* ws = (char*)d_ws;
  const size_t SZ = (size_t)BB * CC * PP * sizeof(float);  // 128 MiB per BCHW f32
  float*    vbuf = (float*)(ws);
  float*    xw   = (float*)(ws + SZ);                 // later reused as csA
  _Float16* gq   = (_Float16*)(ws + 2 * SZ);          // gq+gk region later csB
  _Float16* gk   = (_Float16*)(ws + 2 * SZ + SZ / 2);
  _Float16* gvt  = (_Float16*)(ws + 3 * SZ);
  float*    aout = (float*)(ws + 3 * SZ + SZ / 2);
  _Float16* gxw  = (_Float16*)aout;                   // dead before attn writes aout
  char* sm = ws + 4 * SZ + SZ / 2;
  float* f     = (float*)sm; sm += (size_t)BB * 32 * PP * sizeof(float);
  float* offs  = (float*)sm; sm += (size_t)BB * 2 * PP * sizeof(float);
  float* mchan = (float*)sm; sm += (size_t)BB * PP * sizeof(float);
  float* sab   = (float*)sm; sm += (size_t)BB * PP * sizeof(float);
  float* fmean = (float*)sm; sm += BB * 32 * sizeof(float);
  float* cab   = (float*)sm; sm += BB * 128 * sizeof(float);
  float* maskb = (float*)sm; sm += BB * 256 * sizeof(float);
  float* csA = xw;                     // free after projections
  float* csB = (float*)(ws + 2 * SZ);  // free after attention

  const dim3 blk256(256), blk64(64);
  const dim3 gGemm(PP / 64, BB);
  const int ELT_BLOCKS = (BB * CC * PP) / 256;  // 131072
  const int PIX_BLOCKS = (BB * PP) / 256;       // 1024

  // 1) v = pv(x)
  gemm128_kernel<<<gGemm, blk256, 0, stream>>>(pv_w, pv_b, x, vbuf);
  // 2) f = leaky(rin(cond))
  rin_kernel<<<dim3(PP / 16, BB), blk64, 0, stream>>>(rin_w, rin_b, vbuf, cg, f);
  // 3) offsets + channel mean
  offs_kernel<<<PIX_BLOCKS, blk256, 0, stream>>>(f, ro1_w, ro1_b, ro2_w, ro2_b, offs, mchan);
  // 4) channel attention
  fmean_kernel<<<BB * 32, blk256, 0, stream>>>(f, fmean);
  ca_kernel<<<2, blk256, 0, stream>>>(fmean, rca_w, rca_b, cab);
  // 5) spatial attention
  sa_kernel<<<PIX_BLOCKS, blk256, 0, stream>>>(f, rsa_w, rsa_b, sab);
  // 6) window mask
  mask_kernel<<<4, blk256, 0, stream>>>(mchan, gum, rm1_w, rm1_b, rm2_w, rm2_b, maskb);
  // 7) flow warp
  warp_kernel<<<PIX_BLOCKS, blk256, 0, stream>>>(x, offs, xw);
  // 8) stage masked windows (f16) + fused q/k projections
  stage_kernel<<<ELT_BLOCKS, blk256, 0, stream>>>(xw, vbuf, maskb, gxw, gvt);
  projqk_kernel<<<BB * 256, blk256, 0, stream>>>(gxw, pq_w, pq_b, pk_w, pk_b, gq, gk);
  // 9) attention + residual v*sa*(1-mask)
  attn_kernel<<<BB * 256, blk256, 0, stream>>>(gq, gk, gvt, vbuf, sab, maskb, aout);
  // 10) cs chain: cs1 (1x1) -> dw5 pad2 -> dw5 pad6 dil3
  gemm128_kernel<<<gGemm, blk256, 0, stream>>>(cs1_w, cs1_b, aout, csA);
  dw5_kernel<<<ELT_BLOCKS, blk256, 0, stream>>>(csA, cs2_w, cs2_b, csB, 2, 1);
  dw5_kernel<<<ELT_BLOCKS, blk256, 0, stream>>>(csB, cs3_w, cs3_b, csA, 6, 3);
  // 11) gelu * ca + residual
  out2_kernel<<<ELT_BLOCKS, blk256, 0, stream>>>(csA, aout, cab, csB);
  // 12) final projection
  gemm128_kernel<<<gGemm, blk256, 0, stream>>>(po_w, po_b, csB, (float*)d_out);
}
